// StageZeroSllrcAttention_44358422233479
// MI455X (gfx1250) — matrix operand order repaired and checked by reading, not compiled, not run on hardware
//
#include <hip/hip_runtime.h>
#include <hip/hip_bf16.h>
#include <math.h>

#define B_   4
#define N_   2048
#define D_   768
#define H_   12
#define DPH_ 64

typedef _Float16 v16h __attribute__((ext_vector_type(16)));
typedef _Float16 v8h  __attribute__((ext_vector_type(8)));
typedef float    v8f  __attribute__((ext_vector_type(8)));

// ---------- WMMA fragment helpers (CDNA5 wave32 layouts, ISA 7.12.2) ----------

__device__ __forceinline__ v16h load_frag_pair(const _Float16* p0, const _Float16* p1) {
    v8h lo = *(const v8h*)p0;
    v8h hi = *(const v8h*)p1;
    v16h r;
#pragma unroll
    for (int i = 0; i < 8; ++i) { r[i] = lo[i]; r[i + 8] = hi[i]; }
    return r;
}

// A fragment: 16x32 f16 tile, row-major source, leading dim ld (elements).
// Lane L: row = L&15; halves 0-7 = K (L>>4)*8 .. +7 ; halves 8-15 = K +16 .. +23.
__device__ __forceinline__ v16h load_a_frag(const _Float16* base, int ld, int lane) {
    const int m  = lane & 15;
    const int kh = (lane >> 4) << 3;       // 0 or 8
    const _Float16* p = base + (size_t)m * ld + kh;
    return load_frag_pair(p, p + 16);
}

// B fragment: 32x16 f16 tile, source stored N-major: element (k,n) at n*ldt + k.
// Lane L: col n = L&15; halves 0-7 = K (L>>4)*8 .. +7 ; halves 8-15 = K +16 .. +23.
__device__ __forceinline__ v16h load_b_frag(const _Float16* baseT, int ldt, int lane) {
    const int n  = lane & 15;
    const int kb = (lane >> 4) << 3;       // 0 or 8
    const _Float16* p = baseT + (size_t)n * ldt + kb;
    return load_frag_pair(p, p + 16);
}

__device__ __forceinline__ v8f wmma_f16(v16h a, v16h b, v8f c) {
    return __builtin_amdgcn_wmma_f32_16x16x32_f16(false, a, false, b, (short)0, c, false, false);
}

// one fragment set for a 32x64 wave tile (2 A frags + 4 B frags)
struct FragSet {
    v16h a0, a1, bf0, bf1, bf2, bf3;
};

__device__ __forceinline__ void load_set(FragSet& f, const _Float16* A, const _Float16* Bt,
                                         int lda, int ldb, int k0, int lane) {
    f.a0  = load_a_frag(A + k0,                    lda, lane);
    f.a1  = load_a_frag(A + (size_t)16 * lda + k0, lda, lane);
    f.bf0 = load_b_frag(Bt + (size_t)0  * ldb + k0, ldb, lane);
    f.bf1 = load_b_frag(Bt + (size_t)16 * ldb + k0, ldb, lane);
    f.bf2 = load_b_frag(Bt + (size_t)32 * ldb + k0, ldb, lane);
    f.bf3 = load_b_frag(Bt + (size_t)48 * ldb + k0, ldb, lane);
}

__device__ __forceinline__ void compute_set(const FragSet& f, v8f acc[2][4]) {
    acc[0][0] = wmma_f16(f.a0, f.bf0, acc[0][0]);
    acc[1][0] = wmma_f16(f.a1, f.bf0, acc[1][0]);
    acc[0][1] = wmma_f16(f.a0, f.bf1, acc[0][1]);
    acc[1][1] = wmma_f16(f.a1, f.bf1, acc[1][1]);
    acc[0][2] = wmma_f16(f.a0, f.bf2, acc[0][2]);
    acc[1][2] = wmma_f16(f.a1, f.bf2, acc[1][2]);
    acc[0][3] = wmma_f16(f.a0, f.bf3, acc[0][3]);
    acc[1][3] = wmma_f16(f.a1, f.bf3, acc[1][3]);
}

// Ping-pong double-buffered K loop over [0, D_) in steps of 32 (D_/32 even).
// No register rotation: buf0/buf1 alternate roles in a 2x-unrolled steady state.
__device__ __forceinline__ void gemm_k_loop(v8f acc[2][4], const _Float16* A,
                                            const _Float16* Bt, int lda, int ldb, int lane) {
    FragSet buf0, buf1;
    load_set(buf0, A, Bt, lda, ldb, 0, lane);                 // step 0
    for (int k0 = 32; k0 < D_ - 64; k0 += 64) {
        load_set(buf1, A, Bt, lda, ldb, k0, lane);            // odd step
        compute_set(buf0, acc);
        load_set(buf0, A, Bt, lda, ldb, k0 + 32, lane);       // next even step
        compute_set(buf1, acc);
    }
    load_set(buf1, A, Bt, lda, ldb, D_ - 32, lane);           // last (odd) step
    compute_set(buf0, acc);                                   // step D_/32 - 2
    compute_set(buf1, acc);                                   // step D_/32 - 1
}

// ---------- precision conversion kernels ----------

__global__ void cvt_f32_f16(const float* __restrict__ s, _Float16* __restrict__ d, int n) {
    int i = blockIdx.x * blockDim.x + threadIdx.x;
    if (i < n) d[i] = (_Float16)s[i];
}

// src [G, rows, cols] row-major -> dst [G, cols, rows] (transposed per group) in f16
__global__ void cvtT_f32_f16(const float* __restrict__ s, _Float16* __restrict__ d,
                             int rows, int cols) {
    int g = blockIdx.y;
    const float* sp = s + (size_t)g * rows * cols;
    _Float16*    dp = d + (size_t)g * rows * cols;
    int i = blockIdx.x * blockDim.x + threadIdx.x;
    if (i < rows * cols) {
        int r = i / cols, c = i % cols;
        dp[(size_t)c * rows + r] = (_Float16)sp[i];
    }
}

// ---------- QKV projection: [N,D] @ [D,64] + bias, per (b,h) ----------
// Wave tile: 32 rows x 64 cols, ping-pong pipelined K loop.
// mode 0: Q (scale 1/sqrt(64), row-major out [B,H,N,64])
// mode 1: K (row-major out [B,H,N,64])
// mode 2: V (transposed out [B,H,64,N])
__global__ __launch_bounds__(128) void qkv_kernel(const _Float16* __restrict__ xh,
                                                  const _Float16* __restrict__ wt,
                                                  const float* __restrict__ bias,
                                                  _Float16* __restrict__ dst, int mode) {
    const int lane = threadIdx.x & 31;
    const int w    = threadIdx.x >> 5;
    const int b    = blockIdx.y / H_;
    const int h    = blockIdx.y % H_;
    const int row0 = blockIdx.x * 128 + w * 32;

    const _Float16* A  = xh + ((size_t)b * N_ + row0) * D_;
    const _Float16* Wt = wt + (size_t)h * DPH_ * D_;   // [64(out), D(in)] N-major

    v8f acc[2][4];
#pragma unroll
    for (int u = 0; u < 2; ++u)
#pragma unroll
        for (int t = 0; t < 4; ++t)
#pragma unroll
            for (int r = 0; r < 8; ++r) acc[u][t][r] = 0.0f;

    gemm_k_loop(acc, A, Wt, D_, D_, lane);

    const int   col   = lane & 15;
    const int   rbase = (lane >> 4) * 8;
    const float scale = (mode == 0) ? 0.125f : 1.0f;   // fold 1/sqrt(DPH) into Q
#pragma unroll
    for (int u = 0; u < 2; ++u)
#pragma unroll
        for (int t = 0; t < 4; ++t) {
            float bval = bias[h * DPH_ + t * 16 + col];
#pragma unroll
            for (int r = 0; r < 8; ++r) {
                int   row = row0 + u * 16 + rbase + r;
                float v   = (acc[u][t][r] + bval) * scale;
                if (mode == 2)
                    dst[(((size_t)b * H_ + h) * DPH_ + (t * 16 + col)) * N_ + row] = (_Float16)v;
                else
                    dst[(((size_t)b * H_ + h) * N_ + row) * DPH_ + t * 16 + col] = (_Float16)v;
            }
        }
}

// ---------- flash attention: per wave 16 query rows, stream keys in chunks of 32 ----------
__global__ __launch_bounds__(128) void attn_kernel(const _Float16* __restrict__ qh,
                                                   const _Float16* __restrict__ kh,
                                                   const _Float16* __restrict__ vt,
                                                   _Float16* __restrict__ ctx) {
    __shared__ _Float16 ldsP[4][16 * 32];              // per-wave P staging tile
    const int lane = threadIdx.x & 31;
    const int w    = threadIdx.x >> 5;
    const int b    = blockIdx.y / H_;
    const int h    = blockIdx.y % H_;
    const int q0   = blockIdx.x * 64 + w * 16;

    const _Float16* Q = qh + ((size_t)b * H_ + h) * N_ * DPH_;   // [N,64]
    const _Float16* K = kh + ((size_t)b * H_ + h) * N_ * DPH_;   // [N,64]
    const _Float16* V = vt + ((size_t)b * H_ + h) * DPH_ * N_;   // [64,N]
    _Float16* P = ldsP[w];

    v16h aq0 = load_a_frag(Q + (size_t)q0 * DPH_,      DPH_, lane);
    v16h aq1 = load_a_frag(Q + (size_t)q0 * DPH_ + 32, DPH_, lane);

    v8f   o[4];
    float m[8], l[8];
#pragma unroll
    for (int t = 0; t < 4; ++t)
#pragma unroll
        for (int r = 0; r < 8; ++r) o[t][r] = 0.0f;
#pragma unroll
    for (int r = 0; r < 8; ++r) { m[r] = -1e30f; l[r] = 0.0f; }

    const int col = lane & 15;

    for (int j0 = 0; j0 < N_; j0 += 32) {
        v8f c0, c1;
#pragma unroll
        for (int r = 0; r < 8; ++r) { c0[r] = 0.0f; c1[r] = 0.0f; }

        // S(16x32) = Q(16x64) @ K_chunk^T  (all 4 B frags issued before the WMMAs)
        v16h b0 = load_b_frag(K + (size_t)j0 * DPH_,             DPH_, lane);
        v16h b1 = load_b_frag(K + (size_t)(j0 + 16) * DPH_,      DPH_, lane);
        v16h b2 = load_b_frag(K + (size_t)j0 * DPH_ + 32,        DPH_, lane);
        v16h b3 = load_b_frag(K + (size_t)(j0 + 16) * DPH_ + 32, DPH_, lane);
        c0 = wmma_f16(aq0, b0, c0);
        c1 = wmma_f16(aq0, b1, c1);
        c0 = wmma_f16(aq1, b2, c0);
        c1 = wmma_f16(aq1, b3, c1);

        // issue V fragment loads now: the softmax VALU block below hides their latency
        v16h bv0 = load_b_frag(V + (size_t)0  * N_ + j0, N_, lane);
        v16h bv1 = load_b_frag(V + (size_t)16 * N_ + j0, N_, lane);
        v16h bv2 = load_b_frag(V + (size_t)32 * N_ + j0, N_, lane);
        v16h bv3 = load_b_frag(V + (size_t)48 * N_ + j0, N_, lane);

        // online softmax over the 32 keys of this chunk (rows split across 16-lane halves)
#pragma unroll
        for (int r = 0; r < 8; ++r) {
            float rowmax = fmaxf(c0[r], c1[r]);
#pragma unroll
            for (int s = 1; s < 16; s <<= 1)
                rowmax = fmaxf(rowmax, __shfl_xor(rowmax, s, 32));
            float mn  = fmaxf(m[r], rowmax);
            float fac = __expf(m[r] - mn);
            m[r] = mn;
            float p0 = __expf(c0[r] - mn);
            float p1 = __expf(c1[r] - mn);
            float rs = p0 + p1;
#pragma unroll
            for (int s = 1; s < 16; s <<= 1)
                rs += __shfl_xor(rs, s, 32);
            l[r] = l[r] * fac + rs;
#pragma unroll
            for (int t = 0; t < 4; ++t) o[t][r] *= fac;
            int row = (lane >> 4) * 8 + r;
            P[row * 32 + col]      = (_Float16)p0;     // C-layout -> row-major LDS
            P[row * 32 + 16 + col] = (_Float16)p1;
        }
        asm volatile("s_wait_dscnt 0" ::: "memory");   // LDS stores visible before frag load

        v16h ap = load_a_frag(P, 32, lane);            // P as A operand (16x32)
        o[0] = wmma_f16(ap, bv0, o[0]);
        o[1] = wmma_f16(ap, bv1, o[1]);
        o[2] = wmma_f16(ap, bv2, o[2]);
        o[3] = wmma_f16(ap, bv3, o[3]);
    }

    // normalize and write ctx in [B,N,H*64] (= concat-heads) layout, f16 for out-proj
    float inv_l[8];
#pragma unroll
    for (int r = 0; r < 8; ++r) inv_l[r] = 1.0f / l[r];
#pragma unroll
    for (int t = 0; t < 4; ++t)
#pragma unroll
        for (int r = 0; r < 8; ++r) {
            int row = q0 + (lane >> 4) * 8 + r;
            ctx[((size_t)b * N_ + row) * D_ + h * DPH_ + t * 16 + col] =
                (_Float16)(o[t][r] * inv_l[r]);
        }
}

// ---------- output projection: [B*N, D] @ [D, D] + bo -> f32 out ----------
// Wave tile 32x64, ping-pong pipelined K loop.
__global__ __launch_bounds__(128) void proj_kernel(const _Float16* __restrict__ ctx,
                                                   const _Float16* __restrict__ wot,
                                                   const float* __restrict__ bo,
                                                   float* __restrict__ out) {
    const int lane = threadIdx.x & 31;
    const int w    = threadIdx.x >> 5;
    const int row0 = blockIdx.x * 128 + w * 32;
    const int n0   = blockIdx.y * 64;

    v8f acc[2][4];
#pragma unroll
    for (int u = 0; u < 2; ++u)
#pragma unroll
        for (int t = 0; t < 4; ++t)
#pragma unroll
            for (int r = 0; r < 8; ++r) acc[u][t][r] = 0.0f;

    const _Float16* A  = ctx + (size_t)row0 * D_;
    const _Float16* Bt = wot + (size_t)n0 * D_;

    gemm_k_loop(acc, A, Bt, D_, D_, lane);

    const int col   = lane & 15;
    const int rbase = (lane >> 4) * 8;
#pragma unroll
    for (int u = 0; u < 2; ++u)
#pragma unroll
        for (int t = 0; t < 4; ++t) {
            float bias = bo[n0 + t * 16 + col];
#pragma unroll
            for (int r = 0; r < 8; ++r)
                out[(size_t)(row0 + u * 16 + rbase + r) * D_ + n0 + t * 16 + col] =
                    acc[u][t][r] + bias;
        }
}

// ---------- host side ----------

extern "C" void kernel_launch(void* const* d_in, const int* in_sizes, int n_in,
                              void* d_out, int out_size, void* d_ws, size_t ws_size,
                              hipStream_t stream) {
    const float* x  = (const float*)d_in[0];
    const float* Wq = (const float*)d_in[1];
    const float* bq = (const float*)d_in[2];
    const float* Wk = (const float*)d_in[3];
    const float* bk = (const float*)d_in[4];
    const float* Wv = (const float*)d_in[5];
    const float* bv = (const float*)d_in[6];
    const float* Wo = (const float*)d_in[7];
    const float* bo = (const float*)d_in[8];
    float* out = (float*)d_out;

    const size_t XH  = (size_t)B_ * N_ * D_;        // 6,291,456 halves
    const size_t WT  = (size_t)H_ * DPH_ * D_;      //   589,824 halves
    const size_t WOT = (size_t)D_ * D_;

    _Float16* p     = (_Float16*)d_ws;
    _Float16* x_h   = p; p += XH;
    _Float16* wq_t  = p; p += WT;
    _Float16* wk_t  = p; p += WT;
    _Float16* wv_t  = p; p += WT;
    _Float16* wo_t  = p; p += WOT;
    _Float16* q_h   = p; p += XH;   // [B,H,N,64]
    _Float16* k_h   = p; p += XH;   // [B,H,N,64]
    _Float16* v_t   = p; p += XH;   // [B,H,64,N]
    _Float16* ctx_h = p; p += XH;   // [B,N,D]

    cvt_f32_f16<<<dim3((unsigned)((XH + 255) / 256)), 256, 0, stream>>>(x, x_h, (int)XH);
    cvtT_f32_f16<<<dim3((D_ * DPH_ + 255) / 256, H_), 256, 0, stream>>>(Wq, wq_t, D_, DPH_);
    cvtT_f32_f16<<<dim3((D_ * DPH_ + 255) / 256, H_), 256, 0, stream>>>(Wk, wk_t, D_, DPH_);
    cvtT_f32_f16<<<dim3((D_ * DPH_ + 255) / 256, H_), 256, 0, stream>>>(Wv, wv_t, D_, DPH_);
    cvtT_f32_f16<<<dim3((D_ * D_ + 255) / 256, 1), 256, 0, stream>>>(Wo, wo_t, D_, D_);

    dim3 gq(N_ / 128, B_ * H_);
    qkv_kernel<<<gq, 128, 0, stream>>>(x_h, wq_t, bq, q_h, 0);
    qkv_kernel<<<gq, 128, 0, stream>>>(x_h, wk_t, bk, k_h, 1);
    qkv_kernel<<<gq, 128, 0, stream>>>(x_h, wv_t, bv, v_t, 2);

    attn_kernel<<<dim3(N_ / 64, B_ * H_), 128, 0, stream>>>(q_h, k_h, v_t, ctx_h);

    proj_kernel<<<dim3((B_ * N_) / 128, D_ / 64), 128, 0, stream>>>(ctx_h, wo_t, bo, out);
}
